// AttentionLayer_81475529605500
// MI455X (gfx1250) — matrix operand order repaired and checked by reading, not compiled, not run on hardware
//
#include <hip/hip_runtime.h>
#include <hip/hip_bf16.h>
#include <stdint.h>

typedef __attribute__((ext_vector_type(16))) _Float16 v16h;
typedef __attribute__((ext_vector_type(8)))  _Float16 v8h;
typedef __attribute__((ext_vector_type(4)))  _Float16 v4h;
typedef __attribute__((ext_vector_type(8)))  float    v8f;
typedef __attribute__((ext_vector_type(4)))  unsigned uv4;
typedef __attribute__((ext_vector_type(8)))  int      iv8;
typedef __attribute__((ext_vector_type(4)))  int      iv4;

#define WMMA_F16(A, B, C) \
  __builtin_amdgcn_wmma_f32_16x16x32_f16(false, (A), false, (B), (short)0, (C), false, false)

// Problem constants
constexpr int Bb  = 2;
constexpr int Ss  = 2048;
constexpr int Dd  = 768;
constexpr int Hh  = 12;
constexpr int DHd = 64;

constexpr int MT     = (Bb * Ss) / 16;  // 256 row tiles of x
constexpr int NT     = Dd / 16;         // 48 col tiles of W
constexpr int KSTEPS = Dd / 32;         // 24 k-steps of 32
constexpr int NITER  = Ss / 32;         // 64 key tiles in flash loop

__device__ __forceinline__ unsigned pkh2(float a, float b) {
  union { _Float16 h[2]; unsigned u; } x;
  x.h[0] = (_Float16)a; x.h[1] = (_Float16)b;
  return x.u;
}

__device__ __forceinline__ unsigned lds_byte_off(const void* p) {
  // generic (flat) address of an LDS object: low 32 bits are the LDS byte address
  return (unsigned)(size_t)p;
}

// ---------------------------------------------------------------------------
// TDM: 2D tile load Global -> LDS (f16 elements).  Builds the Tensor DMA
// descriptor (D#) per CDNA5 ISA ch.8 and issues tensor_load_to_lds.
//   group0: [1:0]=count=1, [63:32]=lds_addr, [120:64]=global_addr, [127:126]=2
//   group1: [17:16]=data_size(1 => 2 bytes), tensor_dim0/1, tile_dim0/1,
//           tensor_dim0_stride (all in element units)
// Tracked by TENSORcnt; EXEC is ignored by the TDM.
// This toolchain exposes the 6-arg builtin:
//   (uint32x4 g0, int32x8 g1, int32x4, int32x4, int32x8, i32 cpol)
// ---------------------------------------------------------------------------
__device__ __forceinline__ void tdm_load_tile_2d(unsigned lds_off,
                                                 const _Float16* gptr,
                                                 unsigned tens_d0, unsigned tens_d1,
                                                 unsigned tile_d0, unsigned tile_d1,
                                                 unsigned stride0) {
  const unsigned long long ga = (unsigned long long)(size_t)gptr;
  uv4 g0;
  g0.x = 1u;                                                   // count=1, user D#
  g0.y = lds_off;                                              // lds_addr (bytes)
  g0.z = (unsigned)ga;                                         // global_addr lo
  g0.w = (unsigned)((ga >> 32) & 0x01ffffffu) | (2u << 30);    // addr hi | type=2
  iv8 g1;
  g1[0] = (int)(1u << 16);                                     // data_size=2B, mask=0
  g1[1] = (int)((tens_d0 & 0xffffu) << 16);                    // tensor_dim0 lo16
  g1[2] = (int)(((tens_d0 >> 16) & 0xffffu) |
                ((tens_d1 & 0xffffu) << 16));                  // dim0 hi | dim1 lo
  g1[3] = (int)(((tens_d1 >> 16) & 0xffffu) |
                ((tile_d0 & 0xffffu) << 16));                  // dim1 hi | tile_dim0
  g1[4] = (int)(tile_d1 & 0xffffu);                            // tile_dim1 (dim2=0)
  g1[5] = (int)stride0;                                        // tensor_dim0_stride lo
  g1[6] = 0;                                                   // stride hi | dim1_stride
  g1[7] = 0;
  const iv4 gz4 = {0, 0, 0, 0};                                // 2D: groups 2/3 unused
  const iv8 gz8 = {0, 0, 0, 0, 0, 0, 0, 0};                    // unused hi-dim state
  __builtin_amdgcn_tensor_load_to_lds(g0, g1, gz4, gz4, gz8, 0);
}

// ---------------------------------------------------------------------------
// Pre-pass A: x (fp32) -> xh (f16), same [B*S, D] layout.  4 elems/thread.
// ---------------------------------------------------------------------------
__global__ void cvt_x_f16(const float* __restrict__ x, _Float16* __restrict__ xh) {
  const size_t i = ((size_t)blockIdx.x * blockDim.x + threadIdx.x) * 4;
  const float4 v = *(const float4*)(x + i);
  v4h o; o[0] = (_Float16)v.x; o[1] = (_Float16)v.y; o[2] = (_Float16)v.z; o[3] = (_Float16)v.w;
  *(v4h*)(xh + i) = o;
}

// ---------------------------------------------------------------------------
// Pre-pass B: W[d][n] fp32 -> Wt[n][d] f16 (coalesced via LDS 32x32 tile).
// Wt rows are exactly the WMMA B-operand fragments for the QKV GEMM.
// ---------------------------------------------------------------------------
__global__ void cvt_w_transpose(const float* __restrict__ Wq,
                                const float* __restrict__ Wk,
                                const float* __restrict__ Wv,
                                _Float16* __restrict__ Wt) {
  __shared__ float tile[32][33];
  const int mat = blockIdx.x / 576;          // 3 mats * (24*24) tiles
  const int rem = blockIdx.x % 576;
  const int d0  = (rem / 24) * 32;
  const int n0  = (rem % 24) * 32;
  const float* W = (mat == 0) ? Wq : (mat == 1) ? Wk : Wv;
  const int c  = threadIdx.x & 31;
  const int r0 = threadIdx.x >> 5;           // 8 rows of threads
#pragma unroll
  for (int rr = 0; rr < 32; rr += 8)
    tile[r0 + rr][c] = W[(size_t)(d0 + r0 + rr) * Dd + n0 + c];
  __syncthreads();
  _Float16* Wo = Wt + (size_t)mat * Dd * Dd;
#pragma unroll
  for (int rr = 0; rr < 32; rr += 8)
    Wo[(size_t)(n0 + r0 + rr) * Dd + d0 + c] = (_Float16)tile[c][r0 + rr];
}

// ---------------------------------------------------------------------------
// Kernel 1: fused QKV projection, pure f16 operands / f32 WMMA accumulate.
// One wave per 16x16 output tile.  Q,K stored f16 [B,H,S,DH]; V stored
// transposed [B,H,DH,S].  Q pre-scaled by 1/sqrt(DH).
// ---------------------------------------------------------------------------
__global__ void qkv_proj_wmma(const _Float16* __restrict__ xh,
                              const _Float16* __restrict__ Wt,
                              const float* __restrict__ bq,
                              const float* __restrict__ bk,
                              const float* __restrict__ bv,
                              _Float16* __restrict__ Qh,
                              _Float16* __restrict__ Kh,
                              _Float16* __restrict__ Vt) {
  const int wave = threadIdx.x >> 5;
  const int lane = threadIdx.x & 31;
  const int l16  = lane & 15;
  const bool hiL = lane >= 16;

  const int g     = blockIdx.x * 8 + wave;
  const int matid = g / (MT * NT);           // 0=Q 1=K 2=V
  const int rem   = g - matid * MT * NT;
  const int mt    = rem / NT;
  const int nt    = rem - mt * NT;

  const float* bias = (matid == 0) ? bq : (matid == 1) ? bk : bv;
  const int n = nt * 16 + l16;

  const _Float16* xr = xh + (size_t)(mt * 16 + l16) * Dd;
  const _Float16* wc = Wt + ((size_t)matid * Dd + n) * Dd;   // row n of Wt[mat]

  v8f acc = {};
  for (int kk = 0; kk < KSTEPS; ++kk) {
    const _Float16* xp = xr + kk * 32 + (hiL ? 16 : 0);
    const v8h a0 = *(const v8h*)xp;
    const v8h a1 = *(const v8h*)(xp + 8);
    const v16h a = __builtin_shufflevector(a0, a1, 0,1,2,3,4,5,6,7,8,9,10,11,12,13,14,15);
    const v16h bm = *(const v16h*)(wc + kk * 32 + (hiL ? 16 : 0));
    acc = WMMA_F16(a, bm, acc);
  }

  const float bb    = bias[n];
  const float scale = (matid == 0) ? 0.125f : 1.0f;
#pragma unroll
  for (int r = 0; r < 8; ++r) acc[r] = (acc[r] + bb) * scale;

  const int h  = n >> 6;
  const int dh = n & 63;
  const int m0 = mt * 16 + (hiL ? 8 : 0);
  const int b  = (mt * 16) >> 11;
  const int s0 = m0 & (Ss - 1);

  if (matid == 2) {
    v8h pk;
#pragma unroll
    for (int r = 0; r < 8; ++r) pk[r] = (_Float16)acc[r];
    *(v8h*)(Vt + (((size_t)b * Hh + h) * DHd + dh) * Ss + s0) = pk;
  } else {
    _Float16* dst = (matid == 0) ? Qh : Kh;
#pragma unroll
    for (int r = 0; r < 8; ++r)
      dst[(((size_t)b * Hh + h) * Ss + (s0 + r)) * DHd + dh] = (_Float16)acc[r];
  }
}

// ---------------------------------------------------------------------------
// Kernel 2: flash attention.  8 waves/WG; wave owns 16 queries; 32 keys/iter.
// K/V tiles are DMA'd into double-buffered LDS by the Tensor Data Mover
// (issued by wave 0; TENSORcnt + workgroup barrier for sync); all 8 waves
// read the shared tile via ds_load_b128.  S^T = K*Q^T so softmax state is
// per-lane, and P repacks into B-operand layout with one shfl_xor(16).
// ---------------------------------------------------------------------------
__global__ void attn_fwd_wmma(const _Float16* __restrict__ Qh,
                              const _Float16* __restrict__ Kh,
                              const _Float16* __restrict__ Vt,
                              const unsigned char* __restrict__ mask,
                              float* __restrict__ out) {
  __shared__ _Float16 lk[2][32 * DHd];   // K tile: [key][dh], pitch 64
  __shared__ _Float16 lv[2][DHd * 32];   // V^T tile: [dh][key], pitch 32

  const int wave = threadIdx.x >> 5;
  const int lane = threadIdx.x & 31;
  const int l16  = lane & 15;
  const bool hiL = lane >= 16;

  const int bh = blockIdx.x >> 4;
  const int qb = blockIdx.x & 15;
  const int b  = bh / Hh;
  const int h  = bh - b * Hh;
  const int q  = qb * 128 + wave * 16 + l16;

  const _Float16* Qbase = Qh + (size_t)bh * Ss * DHd;
  const _Float16* Kbase = Kh + (size_t)bh * Ss * DHd;
  const _Float16* Vbase = Vt + (size_t)bh * DHd * Ss;
  const unsigned char* mp = mask + (size_t)b * Ss;

  // Prime the pipeline: tile 0 -> buffer 0 (wave 0 only; TDM ignores EXEC)
  if (wave == 0) {
    tdm_load_tile_2d(lds_byte_off(&lk[0][0]), Kbase, DHd, Ss, DHd, 32, DHd);
    tdm_load_tile_2d(lds_byte_off(&lv[0][0]), Vbase, Ss, DHd, 32, DHd, Ss);
  }

  // Q^T as B operand, loaded once per wave
  const v16h q0 = *(const v16h*)(Qbase + (size_t)q * DHd + (hiL ? 16 : 0));
  const v16h q1 = *(const v16h*)(Qbase + (size_t)q * DHd + 32 + (hiL ? 16 : 0));

  v8f oac[4];
#pragma unroll
  for (int dt = 0; dt < 4; ++dt)
#pragma unroll
    for (int r = 0; r < 8; ++r) oac[dt][r] = 0.0f;

  float mrun = -1e30f;
  float lrun = 0.0f;

  for (int it = 0; it < NITER; ++it) {
    const int kb = it * 32;
    if (wave == 0) {
      if (it + 1 < NITER) {
        const int nb = kb + 32;
        tdm_load_tile_2d(lds_byte_off(&lk[(it + 1) & 1][0]),
                         Kbase + (size_t)nb * DHd, DHd, Ss, DHd, 32, DHd);
        tdm_load_tile_2d(lds_byte_off(&lv[(it + 1) & 1][0]),
                         Vbase + nb, Ss, DHd, 32, DHd, Ss);
        __builtin_amdgcn_s_wait_tensorcnt(2);   // tile `it` complete
      } else {
        __builtin_amdgcn_s_wait_tensorcnt(0);
      }
    }
    __syncthreads();                            // tile `it` visible to all waves

    const _Float16* lkb = &lk[it & 1][0];
    const _Float16* lvb = &lv[it & 1][0];

    // ---- K tile A-fragments from LDS ----
    const _Float16* kp0 = lkb + (size_t)l16 * DHd + (hiL ? 16 : 0);
    const _Float16* kp1 = lkb + (size_t)(16 + l16) * DHd + (hiL ? 16 : 0);
    v8h ta, tb;
    ta = *(const v8h*)(kp0);       tb = *(const v8h*)(kp0 + 8);
    const v16h a00 = __builtin_shufflevector(ta, tb, 0,1,2,3,4,5,6,7,8,9,10,11,12,13,14,15);
    ta = *(const v8h*)(kp0 + 32);  tb = *(const v8h*)(kp0 + 40);
    const v16h a01 = __builtin_shufflevector(ta, tb, 0,1,2,3,4,5,6,7,8,9,10,11,12,13,14,15);
    ta = *(const v8h*)(kp1);       tb = *(const v8h*)(kp1 + 8);
    const v16h a10 = __builtin_shufflevector(ta, tb, 0,1,2,3,4,5,6,7,8,9,10,11,12,13,14,15);
    ta = *(const v8h*)(kp1 + 32);  tb = *(const v8h*)(kp1 + 40);
    const v16h a11 = __builtin_shufflevector(ta, tb, 0,1,2,3,4,5,6,7,8,9,10,11,12,13,14,15);

    v8f st0 = {}, st1 = {};
    st0 = WMMA_F16(a00, q0, st0);
    st0 = WMMA_F16(a01, q1, st0);
    st1 = WMMA_F16(a10, q0, st1);
    st1 = WMMA_F16(a11, q1, st1);

    // ---- mask: s = m ? s : -100 ----
#pragma unroll
    for (int r = 0; r < 8; ++r) {
      const int k0 = kb + r + (hiL ? 8 : 0);
      if (!mp[k0])      st0[r] = -100.0f;
      if (!mp[k0 + 16]) st1[r] = -100.0f;
    }

    // ---- online softmax (per lane; halves combined with shfl_xor(16)) ----
    float tmax = -1e30f;
#pragma unroll
    for (int r = 0; r < 8; ++r) tmax = fmaxf(tmax, fmaxf(st0[r], st1[r]));
    tmax = fmaxf(tmax, __shfl_xor(tmax, 16, 32));

    const float mnew  = fmaxf(mrun, tmax);
    const float alpha = __expf(mrun - mnew);
    float psum = 0.0f;
#pragma unroll
    for (int r = 0; r < 8; ++r) {
      st0[r] = __expf(st0[r] - mnew); psum += st0[r];
      st1[r] = __expf(st1[r] - mnew); psum += st1[r];
    }
    psum += __shfl_xor(psum, 16, 32);
    lrun = lrun * alpha + psum;
    mrun = mnew;

#pragma unroll
    for (int dt = 0; dt < 4; ++dt)
#pragma unroll
      for (int r = 0; r < 8; ++r) oac[dt][r] *= alpha;

    // ---- pack P to f16 B operand (32 keys x 16 queries) ----
    union { unsigned u[8]; v16h v; } pb;
#pragma unroll
    for (int i = 0; i < 4; ++i) {
      const unsigned apk = pkh2(st0[2 * i], st0[2 * i + 1]);
      const unsigned bpk = pkh2(st1[2 * i], st1[2 * i + 1]);
      const unsigned t   = hiL ? apk : bpk;
      const unsigned rr  = (unsigned)__shfl_xor((int)t, 16, 32);
      if (!hiL) { pb.u[i] = apk; pb.u[4 + i] = rr; }
      else      { pb.u[i] = rr;  pb.u[4 + i] = bpk; }
    }

    // ---- O^T += V^T * P^T (V^T fragments from LDS) ----
#pragma unroll
    for (int dt = 0; dt < 4; ++dt) {
      const _Float16* vp = lvb + (size_t)(dt * 16 + l16) * 32 + (hiL ? 16 : 0);
      const v8h va = *(const v8h*)vp;
      const v8h vb = *(const v8h*)(vp + 8);
      const v16h av = __builtin_shufflevector(va, vb, 0,1,2,3,4,5,6,7,8,9,10,11,12,13,14,15);
      oac[dt] = WMMA_F16(av, pb.v, oac[dt]);
    }

    __syncthreads();   // all waves done with buffer (it&1) before TDM refills it
  }

  // ---- epilogue: O[q][d] = O^T[d][q] / l ----
  const float inv = 1.0f / lrun;
  float* op = out + ((size_t)b * Ss + q) * Dd + h * DHd + (hiL ? 8 : 0);
#pragma unroll
  for (int dt = 0; dt < 4; ++dt)
#pragma unroll
    for (int r = 0; r < 8; ++r) op[dt * 16 + r] = oac[dt][r] * inv;
}

// ---------------------------------------------------------------------------
extern "C" void kernel_launch(void* const* d_in, const int* in_sizes, int n_in,
                              void* d_out, int out_size, void* d_ws, size_t ws_size,
                              hipStream_t stream) {
  const float* x  = (const float*)d_in[0];
  const float* Wq = (const float*)d_in[1];
  const float* bq = (const float*)d_in[2];
  const float* Wk = (const float*)d_in[3];
  const float* bk = (const float*)d_in[4];
  const float* Wv = (const float*)d_in[5];
  const float* bv = (const float*)d_in[6];
  const unsigned char* mask = (const unsigned char*)d_in[7];  // jnp bool -> 1 byte

  const size_t qkv_elems = (size_t)Bb * Hh * Ss * DHd;  // 3,145,728 halves
  const size_t x_elems   = (size_t)Bb * Ss * Dd;        // 3,145,728 halves
  const size_t w_elems   = (size_t)3 * Dd * Dd;         // 1,769,472 halves

  _Float16* Qh = (_Float16*)d_ws;
  _Float16* Kh = Qh + qkv_elems;
  _Float16* Vt = Kh + qkv_elems;
  _Float16* xh = Vt + qkv_elems;
  _Float16* Wt = xh + x_elems;
  (void)w_elems; (void)ws_size; (void)in_sizes; (void)n_in; (void)out_size;

  cvt_x_f16<<<(int)(x_elems / 4 / 256), 256, 0, stream>>>(x, xh);
  cvt_w_transpose<<<3 * 24 * 24, 256, 0, stream>>>(Wq, Wk, Wv, Wt);
  qkv_proj_wmma<<<(3 * MT * NT) / 8, 256, 0, stream>>>(xh, Wt, bq, bk, bv, Qh, Kh, Vt);
  attn_fwd_wmma<<<Bb * Hh * 16, 256, 0, stream>>>(Qh, Kh, Vt, mask, (float*)d_out);
}
